// YogaGCN_51711406244070
// MI455X (gfx1250) — compile-verified, matrix operand order checked by reading, not run on hardware
//
#include <hip/hip_runtime.h>

// ---------------------------------------------------------------------------
// GCN forward on MI455X (gfx1250, wave32).
//   Layers 2-4 GEMM: f16 WMMA 16x16x32, fp32 accum. A/B tiles staged to LDS
//   via gfx1250 async load-to-LDS (ASYNCcnt) when available.
//   Activations kept in f16 (halves GEMM streaming reads; scatter stays fp32).
//   Sparse propagate: float4 gather + f32 atomic scatter (L2-resident).
// ---------------------------------------------------------------------------

#define INC 3
#define HID 128
#define NCLS 4
#define LDH 136   // LDS stride in halves: 272B/row -> 4-bank rotation, conflict-free

typedef __attribute__((ext_vector_type(16))) _Float16 v16h;
typedef __attribute__((ext_vector_type(8)))  _Float16 v8h;
typedef __attribute__((ext_vector_type(4)))  _Float16 v4h;
typedef __attribute__((ext_vector_type(8)))  float    v8f;
typedef __attribute__((ext_vector_type(4)))  int      v4i;

#if defined(__gfx1250__) && __has_builtin(__builtin_amdgcn_global_load_async_to_lds_b128)
#define HAVE_ASYNC_LDS 1
#else
#define HAVE_ASYNC_LDS 0
#endif

#if HAVE_ASYNC_LDS
typedef __attribute__((address_space(1))) v4i* gptr_b128;
typedef __attribute__((address_space(3))) v4i* lptr_b128;
#endif

// 16B global -> LDS copy; async data-mover path on gfx1250.
__device__ __forceinline__ void copy16_to_lds(const _Float16* g, _Float16* l) {
#if HAVE_ASYNC_LDS
    __builtin_amdgcn_global_load_async_to_lds_b128(
        (gptr_b128)(void*)g, (lptr_b128)(void*)l, 0, 0);
#else
    *(v8h*)l = *(const v8h*)g;
#endif
}

__device__ __forceinline__ void wait_lds_copies() {
#if HAVE_ASYNC_LDS
    asm volatile("s_wait_asynccnt 0x0" ::: "memory");
#endif
    __syncthreads();
}

// ---------------- degree / normalization precompute ----------------

__global__ void k_init_deg(float* deg, int n) {
    int i = blockIdx.x * blockDim.x + threadIdx.x;
    if (i < n) deg[i] = 1.0f;                       // self loop
}

__global__ void k_edge_deg(const int* col, float* deg, int e) {
    int i = blockIdx.x * blockDim.x + threadIdx.x;
    if (i < e) atomicAdd(&deg[col[i]], 1.0f);
}

__global__ void k_dinv(float* deg, int n) {         // in-place deg -> d^-1/2
    int i = blockIdx.x * blockDim.x + threadIdx.x;
    if (i < n) { float d = deg[i]; deg[i] = d > 0.0f ? rsqrtf(d) : 0.0f; }
}

__global__ void k_edge_norm(const int* row, const int* col, const float* dinv,
                            float* norm, int e) {
    int i = blockIdx.x * blockDim.x + threadIdx.x;
    if (i < e) norm[i] = dinv[row[i]] * dinv[col[i]];
}

__global__ void k_self_norm(const float* dinv, float* selfn, int n) {
    int i = blockIdx.x * blockDim.x + threadIdx.x;
    if (i < n) { float d = dinv[i]; selfn[i] = d * d; }
}

// ---------------- weight convert: fp32 W[k][n] -> f16 WT[n][k] ----------------

__global__ void k_convW(const float* __restrict__ W2, const float* __restrict__ W3,
                        const float* __restrict__ W4, _Float16* __restrict__ wt) {
    int t = blockIdx.x * blockDim.x + threadIdx.x;
    if (t >= 3 * HID * HID) return;
    int l   = t / (HID * HID);
    int rem = t - l * (HID * HID);
    int n = rem >> 7, k = rem & 127;
    const float* W = (l == 0) ? W2 : ((l == 1) ? W3 : W4);
    wt[l * HID * HID + n * HID + k] = (_Float16)W[k * HID + n];
}

// ---------------- layer-1 GEMM (K=3, memory bound, scalar) ----------------

__global__ void k_gemm_in(const float* __restrict__ x, const float* __restrict__ W1,
                          float* __restrict__ out, int n) {
    int t = blockIdx.x * blockDim.x + threadIdx.x;
    int node = t >> 5, j = (t & 31) << 2;
    if (node >= n) return;
    float x0 = x[node * 3 + 0], x1 = x[node * 3 + 1], x2 = x[node * 3 + 2];
    float4 w0 = *(const float4*)(W1 + 0 * HID + j);
    float4 w1 = *(const float4*)(W1 + 1 * HID + j);
    float4 w2 = *(const float4*)(W1 + 2 * HID + j);
    float4 r;
    r.x = x0 * w0.x + x1 * w1.x + x2 * w2.x;
    r.y = x0 * w0.y + x1 * w1.y + x2 * w2.y;
    r.z = x0 * w0.z + x1 * w1.z + x2 * w2.z;
    r.w = x0 * w0.w + x1 * w1.w + x2 * w2.w;
    *(float4*)(out + (size_t)node * HID + j) = r;
}

// ---------------- WMMA GEMM: C[N,128] = A[N,128](f16) @ W[128,128] ----------------
// 128 threads = 4 wave32; each wave owns a 16-row strip of the 64-row block
// tile, sweeping 8 column tiles; K=128 as 4 x (16x16x32) WMMA steps.

__global__ __launch_bounds__(128) void k_gemm_wmma(
    const _Float16* __restrict__ A, const _Float16* __restrict__ WT,
    float* __restrict__ C, int nRows)
{
    __shared__ __align__(16) _Float16 sA[64 * LDH];
    __shared__ __align__(16) _Float16 sB[HID * LDH];
    const int tid = threadIdx.x;
    const int rowBlock = blockIdx.x * 64;
    const bool fullTile = (rowBlock + 64 <= nRows);

    // Stage A: 64 rows x 128 halves = 1024 x 16B chunks (8 per thread).
    #pragma unroll
    for (int i = 0; i < 8; ++i) {
        int idx = tid + i * 128;
        int r  = idx >> 4;              // tile row 0..63
        int c8 = (idx & 15) << 3;       // half offset 0..120
        int gr = rowBlock + r;
        if (gr >= nRows) gr = nRows - 1;  // clamp: pad rows never stored
        copy16_to_lds(A + (size_t)gr * HID + c8, &sA[r * LDH + c8]);
    }
    // Stage B: 128 rows x 128 halves = 2048 x 16B chunks (16 per thread).
    #pragma unroll
    for (int i = 0; i < 16; ++i) {
        int idx = tid + i * 128;
        int n  = idx >> 4;
        int c8 = (idx & 15) << 3;
        copy16_to_lds(WT + (size_t)n * HID + c8, &sB[n * LDH + c8]);
    }
    wait_lds_copies();

    const int wave  = tid >> 5;
    const int lane  = tid & 31;
    const int l16   = lane & 15;
    const int khalf = lane >> 4;          // lanes 0-15: K-half 0, 16-31: K-half 1
    const int rowA  = wave * 16 + l16;

    for (int nt = 0; nt < 8; ++nt) {
        v8f acc = {};
        const int ncol = nt * 16 + l16;
        #pragma unroll
        for (int kk = 0; kk < 4; ++kk) {
            int kb = kk * 32 + khalf * 8;  // per-lane K base (A/B 16-bit layout)
            v8h alo = *(const v8h*)&sA[rowA * LDH + kb];
            v8h ahi = *(const v8h*)&sA[rowA * LDH + kb + 16];
            v8h blo = *(const v8h*)&sB[ncol * LDH + kb];
            v8h bhi = *(const v8h*)&sB[ncol * LDH + kb + 16];
            v16h a = __builtin_shufflevector(alo, ahi, 0,1,2,3,4,5,6,7,8,9,10,11,12,13,14,15);
            v16h b = __builtin_shufflevector(blo, bhi, 0,1,2,3,4,5,6,7,8,9,10,11,12,13,14,15);
            acc = __builtin_amdgcn_wmma_f32_16x16x32_f16(
                false, a, false, b, (short)0, acc, false, false);
        }
        // C/D layout: vgpr r, lanes 0-15 -> M=r, lanes 16-31 -> M=8+r
        int gr0 = rowBlock + wave * 16 + khalf * 8;
        float* cp = C + (size_t)gr0 * HID + nt * 16 + l16;
        if (fullTile) {                       // uniform branch: no exec juggling
            #pragma unroll
            for (int r = 0; r < 8; ++r) cp[r * HID] = acc[r];
        } else {
            #pragma unroll
            for (int r = 0; r < 8; ++r)
                if (gr0 + r < nRows) cp[r * HID] = acc[r];
        }
    }
}

// ---------------- sparse propagation ----------------

__global__ void k_init_agg(const float* __restrict__ t, const float* __restrict__ selfn,
                           float* __restrict__ agg, int n) {
    int i = blockIdx.x * blockDim.x + threadIdx.x;
    int node = i >> 5, j = (i & 31) << 2;
    if (node >= n) return;
    float w = selfn[node];
    float4 v = *(const float4*)(t + (size_t)node * HID + j);
    float4 r = make_float4(w * v.x, w * v.y, w * v.z, w * v.w);
    *(float4*)(agg + (size_t)node * HID + j) = r;
}

__global__ void k_scatter(const float* __restrict__ t, const int* __restrict__ row,
                          const int* __restrict__ col, const float* __restrict__ norm,
                          float* __restrict__ agg, int e) {
    int i = blockIdx.x * blockDim.x + threadIdx.x;
    int ed = i >> 5, j = (i & 31) << 2;
    if (ed >= e) return;
    int r = row[ed], c = col[ed];
    float w = norm[ed];
    float4 v = *(const float4*)(t + (size_t)r * HID + j);
    float* dst = agg + (size_t)c * HID + j;
    atomicAdd(dst + 0, w * v.x);
    atomicAdd(dst + 1, w * v.y);
    atomicAdd(dst + 2, w * v.z);
    atomicAdd(dst + 3, w * v.w);
}

// bias + ReLU; emit f16 activations for the next WMMA GEMM / pooling.
__global__ void k_finalize(const float* __restrict__ agg, const float* __restrict__ bias,
                           _Float16* __restrict__ h, int n) {
    int i = blockIdx.x * blockDim.x + threadIdx.x;
    int node = i >> 5, j = (i & 31) << 2;
    if (node >= n) return;
    float4 v = *(const float4*)(agg + (size_t)node * HID + j);
    float4 b = *(const float4*)(bias + j);
    v4h r;
    r[0] = (_Float16)fmaxf(v.x + b.x, 0.f);
    r[1] = (_Float16)fmaxf(v.y + b.y, 0.f);
    r[2] = (_Float16)fmaxf(v.z + b.z, 0.f);
    r[3] = (_Float16)fmaxf(v.w + b.w, 0.f);
    *(v4h*)(h + (size_t)node * HID + j) = r;
}

// ---------------- pooling + FC ----------------

__global__ void k_zero(float* p, int n) {
    int i = blockIdx.x * blockDim.x + threadIdx.x;
    if (i < n) p[i] = 0.0f;
}

__global__ void k_pool(const _Float16* __restrict__ h, const int* __restrict__ batch,
                       float* __restrict__ sums, float* __restrict__ cnt, int n) {
    int i = blockIdx.x * blockDim.x + threadIdx.x;
    int node = i >> 5, j = (i & 31) << 2;
    if (node >= n) return;
    int g = batch[node];
    v4h v = *(const v4h*)(h + (size_t)node * HID + j);
    float* dst = sums + (size_t)g * HID + j;
    atomicAdd(dst + 0, (float)v[0]);
    atomicAdd(dst + 1, (float)v[1]);
    atomicAdd(dst + 2, (float)v[2]);
    atomicAdd(dst + 3, (float)v[3]);
    if (j == 0) atomicAdd(&cnt[g], 1.0f);
}

__global__ void k_fc(const float* __restrict__ sums, const float* __restrict__ cnt,
                     const float* __restrict__ Wfc, const float* __restrict__ bfc,
                     float* __restrict__ out, int g) {
    int t = blockIdx.x * blockDim.x + threadIdx.x;
    int gi = t >> 2, c = t & 3;
    if (gi >= g) return;
    float acc = 0.0f;
    #pragma unroll 4
    for (int j = 0; j < HID; ++j)
        acc += sums[(size_t)gi * HID + j] * Wfc[j * NCLS + c];
    float inv = 1.0f / fmaxf(cnt[gi], 1.0f);
    out[gi * NCLS + c] = acc * inv + bfc[c];
}

// ---------------- host launcher ----------------

extern "C" void kernel_launch(void* const* d_in, const int* in_sizes, int n_in,
                              void* d_out, int out_size, void* d_ws, size_t ws_size,
                              hipStream_t stream) {
    const float* x    = (const float*)d_in[0];
    const int*   ei   = (const int*)d_in[1];          // [2,E] flat
    const int*   bat  = (const int*)d_in[2];
    const float* W1   = (const float*)d_in[3];
    const float* b1   = (const float*)d_in[4];
    const float* W2   = (const float*)d_in[5];
    const float* b2   = (const float*)d_in[6];
    const float* W3   = (const float*)d_in[7];
    const float* b3   = (const float*)d_in[8];
    const float* W4   = (const float*)d_in[9];
    const float* b4   = (const float*)d_in[10];
    const float* Wfc  = (const float*)d_in[11];
    const float* bfc  = (const float*)d_in[12];
    float* out = (float*)d_out;

    const int N = in_sizes[0] / INC;
    const int E = in_sizes[1] / 2;
    const int G = out_size / NCLS;

    const int* rowp = ei;          // edge_index[0] = source
    const int* colp = ei + E;      // edge_index[1] = target (aggregation side)

    // workspace carve-up (256B aligned)
    char* ws = (char*)d_ws;
    size_t off = 0;
    auto carve = [&](size_t bytes) -> char* {
        char* p = ws + off;
        off = (off + bytes + 255) & ~(size_t)255;
        return p;
    };
    float*    dinv  = (float*)   carve((size_t)N * 4);
    float*    norm  = (float*)   carve((size_t)E * 4);
    float*    selfn = (float*)   carve((size_t)N * 4);
    float*    bufT  = (float*)   carve((size_t)N * HID * 4);
    float*    bufAg = (float*)   carve((size_t)N * HID * 4);
    _Float16* bufH  = (_Float16*)carve((size_t)N * HID * 2);
    _Float16* wt    = (_Float16*)carve((size_t)3 * HID * HID * 2);
    float*    sums  = (float*)   carve((size_t)G * HID * 4);
    float*    cnt   = (float*)   carve((size_t)G * 4);
    (void)ws_size; (void)n_in;

    const int T = 256;
    auto blk = [&](long long work) { return (int)((work + T - 1) / T); };

    // normalization precompute
    k_init_deg <<<blk(N), T, 0, stream>>>(dinv, N);
    k_edge_deg <<<blk(E), T, 0, stream>>>(colp, dinv, E);
    k_dinv     <<<blk(N), T, 0, stream>>>(dinv, N);
    k_edge_norm<<<blk(E), T, 0, stream>>>(rowp, colp, dinv, norm, E);
    k_self_norm<<<blk(N), T, 0, stream>>>(dinv, selfn, N);
    k_convW    <<<blk(3 * HID * HID), T, 0, stream>>>(W2, W3, W4, wt);

    const long long NT = (long long)N * 32;
    const long long ET = (long long)E * 32;
    const int gemmBlocks = (N + 63) / 64;

    // layer 1 (K=3 scalar GEMM)
    k_gemm_in  <<<blk(NT), T, 0, stream>>>(x, W1, bufT, N);
    k_init_agg <<<blk(NT), T, 0, stream>>>(bufT, selfn, bufAg, N);
    k_scatter  <<<blk(ET), T, 0, stream>>>(bufT, rowp, colp, norm, bufAg, E);
    k_finalize <<<blk(NT), T, 0, stream>>>(bufAg, b1, bufH, N);

    // layers 2-4 (WMMA GEMM, f16 activations in, fp32 pre-aggregation out)
    const float* biases[3] = { b2, b3, b4 };
    for (int l = 0; l < 3; ++l) {
        k_gemm_wmma<<<gemmBlocks, 128, 0, stream>>>(bufH, wt + (size_t)l * HID * HID, bufT, N);
        k_init_agg <<<blk(NT), T, 0, stream>>>(bufT, selfn, bufAg, N);
        k_scatter  <<<blk(ET), T, 0, stream>>>(bufT, rowp, colp, norm, bufAg, E);
        k_finalize <<<blk(NT), T, 0, stream>>>(bufAg, biases[l], bufH, N);
    }

    // pooling + FC
    k_zero<<<blk((long long)G * HID), T, 0, stream>>>(sums, G * HID);
    k_zero<<<blk(G), T, 0, stream>>>(cnt, G);
    k_pool<<<blk(NT), T, 0, stream>>>(bufH, bat, sums, cnt, N);
    k_fc  <<<blk((long long)G * NCLS), T, 0, stream>>>(sums, cnt, Wfc, bfc, out, G);
}
